// FlashKAN_81338090651884
// MI455X (gfx1250) — compile-verified, matrix operand order checked by reading, not run on hardware
//
#include <hip/hip_runtime.h>
#include <hip/hip_bf16.h>
#include <stdint.h>

typedef __attribute__((ext_vector_type(16))) _Float16 v16h;
typedef __attribute__((ext_vector_type(8)))  _Float16 v8h;
typedef __attribute__((ext_vector_type(8)))  float    v8f;

#define G_GRID   64
#define KORD     4
#define GKDIM    (G_GRID + KORD)      // 68 rows of w per d
#define DDIM     128
#define ODIM     128
#define BDIM     2048
#define KTOT     (DDIM * GKDIM)       // 8704 = 272 * 32
#define NKC      (KTOT / 32)          // 272 K-chunks of 32
#define DCHUNK   16                   // d's staged in LDS per chunk
#define KCHUNK   (DCHUNK * GKDIM)     // 1088
#define KSTEPS   (KCHUNK / 32)        // 34 WMMA steps / chunk (even)
#define NCHUNKS  (DDIM / DCHUNK)      // 8
#define LDSPITCH 1096                 // 1088 + 8 halves pad -> conflict-free b128
#define KSPLIT   4                    // contraction split across grid.y
#define CH_PER_SPLIT (NCHUNKS / KSPLIT)   // 2 chunks per block
#define PACK_ELEMS (NKC * 8 * 32 * 8)     // 557056 u32 (2.12 MiB)
#define PACK_BYTES ((size_t)PACK_ELEMS * 4)

// ---------------------------------------------------------------------------
// Cubic B-spline basis (de Boor, k=4) with analytic clamped-uniform knots.
// t[j] = -1 + h*clamp(j-3, 0, 64), h = 2/64.
// ---------------------------------------------------------------------------
__device__ __forceinline__ float knot_at(int j) {
    int m = j - 3;
    m = m < 0 ? 0 : (m > 64 ? 64 : m);
    return -1.0f + 0.03125f * (float)m;
}

__device__ __forceinline__ void bspline4(float xc, int& cell,
                                         float& P0, float& P1,
                                         float& P2, float& P3) {
    int c = (int)floorf((xc + 1.0f) * 32.0f);     // (x - t0)/h
    c = c < 0 ? 0 : (c > 63 ? 63 : c);
    cell = c;
    const int i = c + 3;                          // knot-span index
    const float ti  = knot_at(i),     tp1 = knot_at(i + 1);
    const float tp2 = knot_at(i + 2), tp3 = knot_at(i + 3);
    const float tm1 = knot_at(i - 1), tm2 = knot_at(i - 2);
    const float l0 = xc - ti,  r0 = tp1 - xc;
    const float l1 = xc - tm1, r1 = tp2 - xc;
    const float l2 = xc - tm2, r2 = tp3 - xc;
    // degree 1
    float t0 = 1.0f / (r0 + l0);
    float N0 = r0 * t0, N1 = l0 * t0;
    // degree 2
    float u0 = N0 / (r0 + l1);
    float M0 = r0 * u0;
    float sv = l1 * u0;
    float u1 = N1 / (r1 + l0);
    float M1 = sv + r1 * u1;
    sv = l0 * u1;
    float M2 = sv;
    // degree 3
    float w0 = M0 / (r0 + l2);
    P0 = r0 * w0; sv = l2 * w0;
    float w1 = M1 / (r1 + l1);
    P1 = sv + r1 * w1; sv = l1 * w1;
    float w2 = M2 / (r2 + l0);
    P2 = sv + r2 * w2; sv = l0 * w2;
    P3 = sv;
}

// ---------------------------------------------------------------------------
// Pre-pack w[g,d,o] (f32) -> f16 B-fragments in exact WMMA lane/VGPR layout.
// pw[(((kc*8)+nc)*32 + lane)*8 + j] = pair {Wt[K], Wt[K+1]} where
// Wt[c, o] = w[c%68, c/68, o] and
//   K = kc*32 + (j>=4 ? 16:0) + (lane>=16 ? 8:0) + (j&3)*2.
// ---------------------------------------------------------------------------
__global__ __launch_bounds__(256) void pack_w_kernel(
        const float* __restrict__ w, uint32_t* __restrict__ pw) {
    const int flat = blockIdx.x * 256 + threadIdx.x;
    const int j    = flat & 7;
    const int lane = (flat >> 3) & 31;
    const int nc   = (flat >> 8) & 7;
    const int kc   = flat >> 11;

    const int K  = kc * 32 + ((j & 4) ? 16 : 0) + ((lane & 16) ? 8 : 0) + (j & 3) * 2;
    const int n  = nc * 16 + (lane & 15);
    const int d0 = K / GKDIM,  g0 = K - d0 * GKDIM;
    const int K1 = K + 1;
    const int d1 = K1 / GKDIM, g1 = K1 - d1 * GKDIM;

    const float v0 = w[((size_t)g0 * DDIM + d0) * ODIM + n];
    const float v1 = w[((size_t)g1 * DDIM + d1) * ODIM + n];

    union { _Float16 h[2]; uint32_t u; } pk;
    pk.h[0] = (_Float16)v0;
    pk.h[1] = (_Float16)v1;
    pw[flat] = pk.u;
}

// ---------------------------------------------------------------------------
// Main kernel: block = (16 rows of B) x (1/4 of the contraction axis).
// grid = (128 M-tiles, 4 K-splits). 8 waves; wave nc owns O columns
// [nc*16, nc*16+16). Dual accumulators break the WMMA RAW chain.
// Partials written to ws; reduced by reduce_kernel (deterministic order).
// ---------------------------------------------------------------------------
__global__ __launch_bounds__(256) void flashkan_wmma_kernel(
        const float* __restrict__ x,
        const uint32_t* __restrict__ pw,
        float* __restrict__ partial) {
    __shared__ _Float16 As[16][LDSPITCH];

    const int tid   = threadIdx.x;
    const int lane  = tid & 31;
    const int nc    = tid >> 5;          // wave id = N tile
    const int b0    = blockIdx.x * 16;
    const int split = blockIdx.y;        // 0..3, owns d in [split*32, +32)

    const int row = tid & 15;            // row this thread builds splines for
    const int dl  = tid >> 4;            // local d (0..15) within chunk

    // A-fragment addressing (16-bit A 16x32 layout):
    //   lanes 0-15  : M=lane,    K = k0+{0..7}  then k0+{16..23}
    //   lanes 16-31 : M=lane-16, K = k0+{8..15} then k0+{24..31}
    const int arow = lane & 15;
    const int koff = (lane >> 4) << 3;   // 0 or 8

    v8f acc0 = {};
    v8f acc1 = {};

    for (int cc = 0; cc < CH_PER_SPLIT; ++cc) {
        const int chunk = split * CH_PER_SPLIT + cc;

        // --- build A tile: each thread exclusively owns As[row][dl*68 .. +67]
        uint32_t* seg = (uint32_t*)&As[row][dl * GKDIM];
        #pragma unroll
        for (int i2 = 0; i2 < GKDIM / 2; ++i2) seg[i2] = 0u;   // 34 dword zeros

        const int d = chunk * DCHUNK + dl;
        const float xc = x[(size_t)(b0 + row) * DDIM + d];
        int cell; float P0, P1, P2, P3;
        bspline4(xc, cell, P0, P1, P2, P3);
        const float sil = xc / (1.0f + __expf(-xc));           // silu base term

        _Float16* s = &As[row][dl * GKDIM];
        s[cell + 0] = (_Float16)P0;
        s[cell + 1] = (_Float16)P1;
        s[cell + 2] = (_Float16)P2;
        s[cell + 3] = (_Float16)P3;
        s[GKDIM - 1] = (_Float16)sil;                          // slot 67 (cell+3 <= 66)

        __syncthreads();

        // --- 34 WMMA K-steps, two independent accumulation streams
        for (int ks = 0; ks < KSTEPS; ks += 2) {
            const int k0 = ks * 32;
            const int kc = chunk * KSTEPS + ks;                // global K chunk

            const v8h lo0 = *(const v8h*)&As[arow][k0 + koff];
            const v8h hi0 = *(const v8h*)&As[arow][k0 + koff + 16];
            const v16h a0 = __builtin_shufflevector(lo0, hi0,
                0, 1, 2, 3, 4, 5, 6, 7, 8, 9, 10, 11, 12, 13, 14, 15);
            const v16h b0v = *(const v16h*)(
                pw + ((((size_t)kc * 8 + nc) * 32 + lane) * 8));

            const v8h lo1 = *(const v8h*)&As[arow][k0 + 32 + koff];
            const v8h hi1 = *(const v8h*)&As[arow][k0 + 32 + koff + 16];
            const v16h a1 = __builtin_shufflevector(lo1, hi1,
                0, 1, 2, 3, 4, 5, 6, 7, 8, 9, 10, 11, 12, 13, 14, 15);
            const v16h b1v = *(const v16h*)(
                pw + (((((size_t)kc + 1) * 8 + nc) * 32 + lane) * 8));

            acc0 = __builtin_amdgcn_wmma_f32_16x16x32_f16(
                false, a0, false, b0v, (short)0, acc0, false, false);
            acc1 = __builtin_amdgcn_wmma_f32_16x16x32_f16(
                false, a1, false, b1v, (short)0, acc1, false, false);
        }

        __syncthreads();                                       // protect LDS reuse
    }

    const v8f acc = acc0 + acc1;

    // --- store C tile (layout: VGPR r -> M=r / M=8+r, N = lane&15)
    const int n  = lane & 15;
    const int mh = (lane >> 4) * 8;
    float* pout = partial + (size_t)split * BDIM * ODIM;
    #pragma unroll
    for (int r = 0; r < 8; ++r) {
        pout[(size_t)(b0 + mh + r) * ODIM + nc * 16 + n] = acc[r];
    }
}

// ---------------------------------------------------------------------------
// Deterministic K-split reduction: out = sum of 4 partials, fixed order.
// ---------------------------------------------------------------------------
__global__ __launch_bounds__(256) void reduce_kernel(
        const float* __restrict__ partial, float* __restrict__ out) {
    const size_t i = ((size_t)blockIdx.x * 256 + threadIdx.x) * 4;
    float4 s = *(const float4*)(partial + i);
    #pragma unroll
    for (int k = 1; k < KSPLIT; ++k) {
        const float4 q = *(const float4*)(partial + (size_t)k * BDIM * ODIM + i);
        s.x += q.x; s.y += q.y; s.z += q.z; s.w += q.w;
    }
    *(float4*)(out + i) = s;
}

// ---------------------------------------------------------------------------
extern "C" void kernel_launch(void* const* d_in, const int* in_sizes, int n_in,
                              void* d_out, int out_size, void* d_ws, size_t ws_size,
                              hipStream_t stream) {
    const float* x = (const float*)d_in[0];   // [2048, 128] f32
    const float* w = (const float*)d_in[1];   // [68, 128, 128] f32
    // d_in[2] = t (knots; recomputed analytically), d_in[3] = k (== 4, static)

    uint32_t* pw      = (uint32_t*)d_ws;                          // 2.12 MiB
    float*    partial = (float*)((char*)d_ws + PACK_BYTES);       // 4 MiB

    pack_w_kernel<<<PACK_ELEMS / 256, 256, 0, stream>>>(w, pw);

    dim3 grid(BDIM / 16, KSPLIT);
    flashkan_wmma_kernel<<<grid, 256, 0, stream>>>(x, pw, partial);

    reduce_kernel<<<(BDIM * ODIM) / (256 * 4), 256, 0, stream>>>(
        partial, (float*)d_out);
}
